// QSLinear_38525856645127
// MI455X (gfx1250) — 8-bit matrix operand order repaired and checked by reading, compiled, not run on hardware
//
#include <hip/hip_runtime.h>


typedef __attribute__((ext_vector_type(8))) int v8i;

#define IN_DIM   1024
#define OUT_DIM  1024
#define ROWS_X   32768
#define SPARSE_K 512

union FragA { v8i v; int2 d[4]; };
union FragB { v8i v; int4 q[2]; };

__global__ __launch_bounds__(256) void quant_w_kernel(
    const float* __restrict__ w, signed char* __restrict__ wq,
    float* __restrict__ wscale)
{
    __shared__ float red[256];
    const int tid = threadIdx.x;
    const int row = blockIdx.x;
    const float4 v = *(const float4*)(w + (size_t)row * IN_DIM + tid * 4);

    float m = fmaxf(fmaxf(fabsf(v.x), fabsf(v.y)), fmaxf(fabsf(v.z), fabsf(v.w)));
    red[tid] = m;
    __syncthreads();
    for (int s = 128; s > 0; s >>= 1) {
        if (tid < s) red[tid] = fmaxf(red[tid], red[tid + s]);
        __syncthreads();
    }
    float scale = red[0] * (1.0f / 127.0f);
    if (scale == 0.0f) scale = 1.0f;
    const float inv = 1.0f / scale;
    if (tid == 0) wscale[row] = scale;

    int qx = (int)rintf(fminf(fmaxf(v.x * inv, -127.f), 127.f));
    int qy = (int)rintf(fminf(fmaxf(v.y * inv, -127.f), 127.f));
    int qz = (int)rintf(fminf(fmaxf(v.z * inv, -127.f), 127.f));
    int qw4 = (int)rintf(fminf(fmaxf(v.w * inv, -127.f), 127.f));
    int packed = (qx & 0xff) | ((qy & 0xff) << 8) | ((qz & 0xff) << 16) | (qw4 << 24);
    ((int*)wq)[(size_t)row * (IN_DIM / 4) + tid] = packed;
}

__global__ __launch_bounds__(256) void quant_x_kernel(
    const float* __restrict__ x, signed char* __restrict__ xq,
    float* __restrict__ xscale)
{
    __shared__ float red[256];
    __shared__ int   hist[128];
    __shared__ int   thr_s;
    const int tid = threadIdx.x;
    const int row = blockIdx.x;
    const float4 v = *(const float4*)(x + (size_t)row * IN_DIM + tid * 4);

    float m = fmaxf(fmaxf(fabsf(v.x), fabsf(v.y)), fmaxf(fabsf(v.z), fabsf(v.w)));
    red[tid] = m;
    if (tid < 128) hist[tid] = 0;
    __syncthreads();
    for (int s = 128; s > 0; s >>= 1) {
        if (tid < s) red[tid] = fmaxf(red[tid], red[tid + s]);
        __syncthreads();
    }
    float scale = red[0] * (1.0f / 127.0f);
    if (scale == 0.0f) scale = 1.0f;
    const float inv = 1.0f / scale;
    if (tid == 0) xscale[row] = scale;

    int q[4];
    q[0] = (int)rintf(fminf(fmaxf(v.x * inv, -127.f), 127.f));
    q[1] = (int)rintf(fminf(fmaxf(v.y * inv, -127.f), 127.f));
    q[2] = (int)rintf(fminf(fmaxf(v.z * inv, -127.f), 127.f));
    q[3] = (int)rintf(fminf(fmaxf(v.w * inv, -127.f), 127.f));
    #pragma unroll
    for (int j = 0; j < 4; ++j) atomicAdd(&hist[abs(q[j])], 1);
    __syncthreads();

    if (tid == 0) {
        int cum = 0, t = 127;
        for (int b = 0; b < 128; ++b) {
            cum += hist[b];
            if (cum >= SPARSE_K) { t = b; break; }
        }
        thr_s = t;
    }
    __syncthreads();
    const int thr = thr_s;

    #pragma unroll
    for (int j = 0; j < 4; ++j)
        if (abs(q[j]) <= thr) q[j] = 0;
    int packed = (q[0] & 0xff) | ((q[1] & 0xff) << 8) | ((q[2] & 0xff) << 16) | (q[3] << 24);
    ((int*)xq)[(size_t)row * (IN_DIM / 4) + tid] = packed;
}

__global__ __launch_bounds__(256) void gemm_i8_kernel(
    const signed char* __restrict__ xq, const signed char* __restrict__ wq,
    const float* __restrict__ xs, const float* __restrict__ ws,
    const float* __restrict__ bias, float* __restrict__ out)
{
    __shared__ __align__(16) signed char As[128 * 64];
    __shared__ __align__(16) signed char Bs[128 * 64];

    const int tid  = threadIdx.x;
    const int lane = tid & 31;
    const int wave = tid >> 5;
    const int lo   = lane & 15;
    const int hi   = lane >> 4;
    const int wm   = wave & 1;
    const int wn   = wave >> 1;

    const int m0 = blockIdx.y * 128;
    const int n0 = blockIdx.x * 128;

    const int cr = tid >> 2;
    const int cc = (tid & 3) * 16;

    v8i acc[4][2] = {};

    for (int k0 = 0; k0 < IN_DIM; k0 += 64) {
        *(int4*)(As + cr * 64 + cc) =
            *(const int4*)(xq + (size_t)(m0 + cr) * IN_DIM + k0 + cc);
        *(int4*)(As + (cr + 64) * 64 + cc) =
            *(const int4*)(xq + (size_t)(m0 + cr + 64) * IN_DIM + k0 + cc);
        *(int4*)(Bs + cr * 64 + cc) =
            *(const int4*)(wq + (size_t)(n0 + cr) * IN_DIM + k0 + cc);
        *(int4*)(Bs + (cr + 64) * 64 + cc) =
            *(const int4*)(wq + (size_t)(n0 + cr + 64) * IN_DIM + k0 + cc);
        if (k0 + 64 < IN_DIM) {
            __builtin_prefetch(xq + (size_t)(m0 + cr) * IN_DIM + k0 + 64 + cc, 0, 0);
            __builtin_prefetch(wq + (size_t)(n0 + cr) * IN_DIM + k0 + 64 + cc, 0, 0);
        }
        __syncthreads();

        FragB b[2];
        #pragma unroll
        for (int nt = 0; nt < 2; ++nt) {
            const int n = wn * 32 + nt * 16 + lo;
            b[nt].q[0] = *(const int4*)(Bs + n * 64 + hi * 16);
            b[nt].q[1] = *(const int4*)(Bs + n * 64 + 32 + hi * 16);
        }
        #pragma unroll
        for (int mt = 0; mt < 4; ++mt) {
            FragA a;
            const int m = wm * 64 + mt * 16 + lo;
            #pragma unroll
            for (int p = 0; p < 4; ++p)
                a.d[p] = *(const int2*)(As + m * 64 + (p >> 1) * 32 + hi * 16 + (p & 1) * 8);
            #pragma unroll
            for (int nt = 0; nt < 2; ++nt)
                acc[mt][nt] = __builtin_amdgcn_wmma_i32_16x16x64_iu8(
                     true, a.v,  true, b[nt].v,
                    acc[mt][nt],  false,  false);
        }
        __syncthreads();
    }

    #pragma unroll
    for (int mt = 0; mt < 4; ++mt) {
        #pragma unroll
        for (int nt = 0; nt < 2; ++nt) {
            const int col = n0 + wn * 32 + nt * 16 + lo;
            const float sw = ws[col];
            const float bv = bias[col];
            #pragma unroll
            for (int r = 0; r < 8; ++r) {
                const int row = m0 + wm * 64 + mt * 16 + hi * 8 + r;
                out[(size_t)row * OUT_DIM + col] =
                    (float)acc[mt][nt][r] * xs[row] * sw + bv;
            }
        }
    }
}

extern "C" void kernel_launch(void* const* d_in, const int* in_sizes, int n_in,
                              void* d_out, int out_size, void* d_ws, size_t ws_size,
                              hipStream_t stream) {
    const float* x    = (const float*)d_in[0];
    const float* w    = (const float*)d_in[1];
    const float* bias = (const float*)d_in[2];
    float*       out  = (float*)d_out;

    char* ws = (char*)d_ws;
    signed char* xq = (signed char*)ws;
    signed char* wq = (signed char*)(ws + (size_t)ROWS_X * IN_DIM);
    float* xscale   = (float*)(ws + (size_t)ROWS_X * IN_DIM + (size_t)OUT_DIM * IN_DIM);
    float* wscale   = xscale + ROWS_X;

    quant_w_kernel<<<OUT_DIM, 256, 0, stream>>>(w, wq, wscale);
    quant_x_kernel<<<ROWS_X, 256, 0, stream>>>(x, xq, xscale);
    gemm_i8_kernel<<<dim3(OUT_DIM / 128, ROWS_X / 128), 256, 0, stream>>>(
        xq, wq, xscale, wscale, bias, out);
}
